// MemoryEfficientAttention_63393717289162
// MI455X (gfx1250) — matrix operand order repaired and checked by reading, not compiled, not run on hardware
//
#include <hip/hip_runtime.h>
#include <hip/hip_bf16.h>

// ---------------------------------------------------------------------------
// MI455X (gfx1250) fused MHA forward.
// B=2, S=2048, D=1024, H=16, Hd=64.  ~69 GFLOP vs ~150MB HBM traffic
// => compute bound => bf16 WMMA everywhere (v_wmma_f32_16x16x32_bf16),
// async global->LDS (b128) for the bf16 A-tile of the output projection,
// softmax folded into base-2 domain (scale*log2e fused into Q projection).
// Workspace (bf16): Qb | Kb | Vt | Ob = 4 * 8MB = 32MB.
// ---------------------------------------------------------------------------

typedef __bf16 bf16;
typedef __bf16 v16bf __attribute__((ext_vector_type(16)));
typedef __bf16 v8bf  __attribute__((ext_vector_type(8)));
typedef __bf16 v4bf  __attribute__((ext_vector_type(4)));
typedef float  v8f   __attribute__((ext_vector_type(8)));

#define S_LEN   2048
#define DMODEL  1024
#define NHEAD   16
#define HDIM    64
// 1/sqrt(64) * log2(e): fold softmax scale + base-2 conversion into Q.
#define QSCALE  0.18033688011112042f

static __device__ __forceinline__ v16bf cat8(v8bf lo, v8bf hi) {
  union U { v16bf v; v8bf h[2]; } u;
  u.h[0] = lo; u.h[1] = hi;
  return u.v;
}

static __device__ __forceinline__ v8f wmma_bf16(v16bf a, v16bf b, v8f c) {
  // D = A(16x32 bf16) * B(32x16 bf16) + C(16x16 f32)
  return __builtin_amdgcn_wmma_f32_16x16x32_bf16(false, a, false, b,
                                                 (short)0, c, false, false);
}

// ---------------------------------------------------------------------------
// Kernel 1: QKV projections.  out = X @ W^T + b   (M=4096, N=1024, K=1024)
// 256 threads (8 waves), block tile 128x64, K-step 64 -> 8 WMMA / barrier pair.
// z==0 -> Qb[b][h][s][hd] (pre-scaled by QSCALE), z==1 -> Kb,
// z==2 -> Vt[b][h][hd][s] (transposed: contiguous b128 store per lane).
// ---------------------------------------------------------------------------
__global__ __launch_bounds__(256) void qkv_proj_kernel(
    const float* __restrict__ Xq, const float* __restrict__ Xk,
    const float* __restrict__ Xv,
    const float* __restrict__ Wq, const float* __restrict__ bq,
    const float* __restrict__ Wk, const float* __restrict__ bk,
    const float* __restrict__ Wv, const float* __restrict__ bv,
    bf16* __restrict__ Qb, bf16* __restrict__ Kb, bf16* __restrict__ Vt)
{
  __shared__ bf16 As[128 * 64];
  __shared__ bf16 Bs[64 * 64];

  const int z = blockIdx.z;
  const float* X    = (z == 0) ? Xq : (z == 1) ? Xk : Xv;
  const float* W    = (z == 0) ? Wq : (z == 1) ? Wk : Wv;
  const float* bias = (z == 0) ? bq : (z == 1) ? bk : bv;

  const int tid  = threadIdx.x;
  const int m0   = blockIdx.y * 128;
  const int n0   = blockIdx.x * 64;
  const int wave = tid >> 5, lane = tid & 31;
  const int wm = wave >> 1, wn = wave & 1;
  const int lh = lane >> 4, lm = lane & 15;

  v8f acc[2][2] = {};

  for (int k0 = 0; k0 < DMODEL; k0 += 64) {
    // stage A (activations) 128x64, f32 -> bf16
#pragma unroll
    for (int i = 0; i < 8; ++i) {
      int e = i * 1024 + tid * 4;
      int row = e >> 6, col = e & 63;
      float4 f = *(const float4*)(X + (size_t)(m0 + row) * DMODEL + k0 + col);
      v4bf hv; hv[0] = (bf16)f.x; hv[1] = (bf16)f.y; hv[2] = (bf16)f.z; hv[3] = (bf16)f.w;
      *(v4bf*)(&As[row * 64 + col]) = hv;
    }
    // stage B (weight rows) 64x64, f32 -> bf16
#pragma unroll
    for (int i = 0; i < 4; ++i) {
      int e = i * 1024 + tid * 4;
      int row = e >> 6, col = e & 63;
      float4 f = *(const float4*)(W + (size_t)(n0 + row) * DMODEL + k0 + col);
      v4bf hv; hv[0] = (bf16)f.x; hv[1] = (bf16)f.y; hv[2] = (bf16)f.z; hv[3] = (bf16)f.w;
      *(v4bf*)(&Bs[row * 64 + col]) = hv;
    }
    __syncthreads();

    v16bf a[2][2], b[2][2];
#pragma unroll
    for (int sm = 0; sm < 2; ++sm) {
      int row = wm * 32 + sm * 16 + lm;
#pragma unroll
      for (int dd = 0; dd < 2; ++dd) {
        v8bf lo = *(const v8bf*)(&As[row * 64 + dd * 32 + lh * 16]);
        v8bf hi = *(const v8bf*)(&As[row * 64 + dd * 32 + lh * 16 + 8]);
        a[sm][dd] = cat8(lo, hi);
      }
    }
#pragma unroll
    for (int sn = 0; sn < 2; ++sn) {
      int row = wn * 32 + sn * 16 + lm;
#pragma unroll
      for (int dd = 0; dd < 2; ++dd) {
        v8bf lo = *(const v8bf*)(&Bs[row * 64 + dd * 32 + lh * 16]);
        v8bf hi = *(const v8bf*)(&Bs[row * 64 + dd * 32 + lh * 16 + 8]);
        b[sn][dd] = cat8(lo, hi);
      }
    }
#pragma unroll
    for (int sm = 0; sm < 2; ++sm)
#pragma unroll
      for (int sn = 0; sn < 2; ++sn)
#pragma unroll
        for (int dd = 0; dd < 2; ++dd)
          acc[sm][sn] = wmma_bf16(a[sm][dd], b[sn][dd], acc[sm][sn]);
    __syncthreads();
  }

  // epilogue: bias (+ QSCALE for Q) + scatter to head layout
#pragma unroll
  for (int sm = 0; sm < 2; ++sm) {
#pragma unroll
    for (int sn = 0; sn < 2; ++sn) {
      int n = n0 + wn * 32 + sn * 16 + lm;
      float bia = bias[n];
      int hh = n >> 6, hd = n & 63;
      if (z < 2) {
        bf16* dst = (z == 0) ? Qb : Kb;
        float mul = (z == 0) ? QSCALE : 1.0f;
#pragma unroll
        for (int r = 0; r < 8; ++r) {
          int m = m0 + wm * 32 + sm * 16 + r + 8 * lh;
          int bb = m >> 11, s = m & (S_LEN - 1);
          dst[((size_t)(bb * NHEAD + hh) * S_LEN + s) * HDIM + hd] =
              (bf16)((acc[sm][sn][r] + bia) * mul);
        }
      } else {
        // V transposed: lane's 8 accumulator rows are 8 consecutive s -> b128
        int mbase = m0 + wm * 32 + sm * 16 + 8 * lh;
        int bb = mbase >> 11, s = mbase & (S_LEN - 1);
        v8bf pack;
#pragma unroll
        for (int r = 0; r < 8; ++r) pack[r] = (bf16)(acc[sm][sn][r] + bia);
        *(v8bf*)(&Vt[((size_t)(bb * NHEAD + hh) * HDIM + hd) * S_LEN + s]) = pack;
      }
    }
  }
}

// ---------------------------------------------------------------------------
// Kernel 2: flash attention, one wave per 16-row query tile per (b,h).
// Q is pre-scaled by scale*log2e, so softmax runs natively in exp2.
// ---------------------------------------------------------------------------
__global__ __launch_bounds__(128) void flash_attn_kernel(
    const bf16* __restrict__ Qb, const bf16* __restrict__ Kb,
    const bf16* __restrict__ Vt, bf16* __restrict__ Ob)
{
  __shared__ bf16 Pl[4 * 16 * 64];  // per-wave P staging (C-frag -> A-frag)

  const int tid  = threadIdx.x;
  const int wave = tid >> 5, lane = tid & 31;
  const int lh = lane >> 4, lm = lane & 15;
  const int T  = blockIdx.x * 4 + wave;   // global query-tile id
  const int bh = T >> 7;                  // 128 tiles of 16 rows per (b,h)
  const int qt = T & 127;
  const int b  = bh >> 4, h = bh & 15;
  bf16* wP = &Pl[wave * (16 * 64)];

  // Q A-fragments for K-dim 64 (two 16x16x32 steps), held in registers
  const bf16* qrow = Qb + ((size_t)bh * S_LEN + qt * 16 + lm) * HDIM;
  v16bf qa[2];
#pragma unroll
  for (int dd = 0; dd < 2; ++dd) {
    v8bf lo = *(const v8bf*)(qrow + dd * 32 + lh * 16);
    v8bf hi = *(const v8bf*)(qrow + dd * 32 + lh * 16 + 8);
    qa[dd] = cat8(lo, hi);
  }

  v8f o[4] = {};
  float mrow[8], lrow[8];
#pragma unroll
  for (int r = 0; r < 8; ++r) { mrow[r] = -1e30f; lrow[r] = 0.f; }

  for (int j0 = 0; j0 < S_LEN; j0 += 64) {
    if (j0 + 64 < S_LEN) {  // prefetch next key block (global_prefetch_b8)
      __builtin_prefetch(Kb + ((size_t)bh * S_LEN + j0 + 64 + lane) * HDIM, 0, 1);
      __builtin_prefetch(Vt + ((size_t)bh * HDIM + lane) * S_LEN + j0 + 64, 0, 1);
    }

    // ---- X = (Q*c) K^T for 4 key tiles of 16 (already in log2 domain) --
    v8f sc[4] = {};
#pragma unroll
    for (int kt = 0; kt < 4; ++kt) {
      const bf16* krow = Kb + ((size_t)bh * S_LEN + j0 + kt * 16 + lm) * HDIM;
#pragma unroll
      for (int dd = 0; dd < 2; ++dd) {
        v8bf lo = *(const v8bf*)(krow + dd * 32 + lh * 16);
        v8bf hi = *(const v8bf*)(krow + dd * 32 + lh * 16 + 8);
        sc[kt] = wmma_bf16(qa[dd], cat8(lo, hi), sc[kt]);
      }
    }

    // ---- online softmax (base 2) ---------------------------------------
    float mnew[8], alpha[8];
#pragma unroll
    for (int r = 0; r < 8; ++r) {
      float v = fmaxf(fmaxf(sc[0][r], sc[1][r]), fmaxf(sc[2][r], sc[3][r]));
#pragma unroll
      for (int msk = 1; msk < 16; msk <<= 1)
        v = fmaxf(v, __shfl_xor(v, msk, 32));   // row reduce within 16-lane half
      mnew[r]  = fmaxf(mrow[r], v);
      alpha[r] = exp2f(mrow[r] - mnew[r]);
      mrow[r]  = mnew[r];
    }
#pragma unroll
    for (int r = 0; r < 8; ++r) {
      float s0 = 0.f;
#pragma unroll
      for (int kt = 0; kt < 4; ++kt) {
        float p = exp2f(sc[kt][r] - mnew[r]);
        sc[kt][r] = p;
        s0 += p;
      }
#pragma unroll
      for (int msk = 1; msk < 16; msk <<= 1)
        s0 += __shfl_xor(s0, msk, 32);
      lrow[r] = lrow[r] * alpha[r] + s0;
    }
#pragma unroll
    for (int dt = 0; dt < 4; ++dt)
#pragma unroll
      for (int r = 0; r < 8; ++r) o[dt][r] *= alpha[r];

    // ---- P: C-layout -> A-layout through per-wave LDS ------------------
#pragma unroll
    for (int kt = 0; kt < 4; ++kt)
#pragma unroll
      for (int r = 0; r < 8; ++r)
        wP[(r + 8 * lh) * 64 + kt * 16 + lm] = (bf16)sc[kt][r];
    asm volatile("s_wait_dscnt 0x0" ::: "memory");  // cross-lane LDS RAW

    // ---- O += P V ------------------------------------------------------
#pragma unroll
    for (int kk = 0; kk < 2; ++kk) {
      v8bf plo = *(const v8bf*)(wP + lm * 64 + kk * 32 + lh * 16);
      v8bf phi = *(const v8bf*)(wP + lm * 64 + kk * 32 + lh * 16 + 8);
      v16bf pa = cat8(plo, phi);
#pragma unroll
      for (int dt = 0; dt < 4; ++dt) {
        const bf16* vrow =
            Vt + ((size_t)bh * HDIM + dt * 16 + lm) * S_LEN + j0 + kk * 32 + lh * 16;
        v8bf lo = *(const v8bf*)(vrow);
        v8bf hi = *(const v8bf*)(vrow + 8);
        o[dt] = wmma_bf16(pa, cat8(lo, hi), o[dt]);
      }
    }
  }

  // ---- normalize and write O back to [B][S][D] bf16 --------------------
#pragma unroll
  for (int dt = 0; dt < 4; ++dt) {
#pragma unroll
    for (int r = 0; r < 8; ++r) {
      int s   = qt * 16 + r + 8 * lh;
      int col = h * HDIM + dt * 16 + lm;
      Ob[((size_t)b * S_LEN + s) * DMODEL + col] = (bf16)(o[dt][r] / lrow[r]);
    }
  }
}

// ---------------------------------------------------------------------------
// Kernel 3: output projection.  out(f32) = Ob(bf16) @ Wo^T + bo
// A tile is already bf16 -> copy global->LDS with CDNA5 async-LDS b128 loads
// (no VGPR round trip, tracked by ASYNCcnt).
// ---------------------------------------------------------------------------
__global__ __launch_bounds__(256) void out_proj_kernel(
    const bf16* __restrict__ Ob, const float* __restrict__ Wo,
    const float* __restrict__ bo, float* __restrict__ out)
{
  __shared__ bf16 As[128 * 64];
  __shared__ bf16 Bs[64 * 64];

  const int tid  = threadIdx.x;
  const int m0   = blockIdx.y * 128;
  const int n0   = blockIdx.x * 64;
  const int wave = tid >> 5, lane = tid & 31;
  const int wm = wave >> 1, wn = wave & 1;
  const int lh = lane >> 4, lm = lane & 15;

  const unsigned ldsA = (unsigned)(uintptr_t)(&As[0]);  // LDS byte offset

  v8f acc[2][2] = {};

  for (int k0 = 0; k0 < DMODEL; k0 += 64) {
    // A tile 128x64 bf16 = 16KB: 4 rounds of async b128 copies (16B/lane)
#pragma unroll
    for (int i = 0; i < 4; ++i) {
      int e = i * 2048 + tid * 8;          // element index in tile
      int row = e >> 6, col = e & 63;
      const bf16* src = Ob + (size_t)(m0 + row) * DMODEL + k0 + col;
      unsigned dstoff = ldsA + (unsigned)e * 2;
      asm volatile("global_load_async_to_lds_b128 %0, %1, off"
                   :: "v"(dstoff), "v"((unsigned long long)(uintptr_t)src)
                   : "memory");
    }
    // B (weight rows) 64x64, f32 -> bf16 via VALU (conversion needed)
#pragma unroll
    for (int i = 0; i < 4; ++i) {
      int e = i * 1024 + tid * 4;
      int row = e >> 6, col = e & 63;
      float4 f = *(const float4*)(Wo + (size_t)(n0 + row) * DMODEL + k0 + col);
      v4bf hv; hv[0] = (bf16)f.x; hv[1] = (bf16)f.y; hv[2] = (bf16)f.z; hv[3] = (bf16)f.w;
      *(v4bf*)(&Bs[row * 64 + col]) = hv;
    }
    asm volatile("s_wait_asynccnt 0x0" ::: "memory");   // my async copies landed
    __syncthreads();                                    // everyone's landed

    v16bf a[2][2], b[2][2];
#pragma unroll
    for (int sm = 0; sm < 2; ++sm) {
      int row = wm * 32 + sm * 16 + lm;
#pragma unroll
      for (int dd = 0; dd < 2; ++dd) {
        v8bf lo = *(const v8bf*)(&As[row * 64 + dd * 32 + lh * 16]);
        v8bf hi = *(const v8bf*)(&As[row * 64 + dd * 32 + lh * 16 + 8]);
        a[sm][dd] = cat8(lo, hi);
      }
    }
#pragma unroll
    for (int sn = 0; sn < 2; ++sn) {
      int row = wn * 32 + sn * 16 + lm;
#pragma unroll
      for (int dd = 0; dd < 2; ++dd) {
        v8bf lo = *(const v8bf*)(&Bs[row * 64 + dd * 32 + lh * 16]);
        v8bf hi = *(const v8bf*)(&Bs[row * 64 + dd * 32 + lh * 16 + 8]);
        b[sn][dd] = cat8(lo, hi);
      }
    }
#pragma unroll
    for (int sm = 0; sm < 2; ++sm)
#pragma unroll
      for (int sn = 0; sn < 2; ++sn)
#pragma unroll
        for (int dd = 0; dd < 2; ++dd)
          acc[sm][sn] = wmma_bf16(a[sm][dd], b[sn][dd], acc[sm][sn]);
    __syncthreads();
  }

#pragma unroll
  for (int sm = 0; sm < 2; ++sm) {
#pragma unroll
    for (int sn = 0; sn < 2; ++sn) {
      int n = n0 + wn * 32 + sn * 16 + lm;
      float bia = bo[n];
#pragma unroll
      for (int r = 0; r < 8; ++r) {
        int m = m0 + wm * 32 + sm * 16 + r + 8 * lh;
        out[(size_t)m * DMODEL + n] = acc[sm][sn][r] + bia;
      }
    }
  }
}

// ---------------------------------------------------------------------------
extern "C" void kernel_launch(void* const* d_in, const int* in_sizes, int n_in,
                              void* d_out, int out_size, void* d_ws, size_t ws_size,
                              hipStream_t stream) {
  const float* q  = (const float*)d_in[0];
  const float* k  = (const float*)d_in[1];
  const float* v  = (const float*)d_in[2];
  const float* Wq = (const float*)d_in[3];
  const float* bq = (const float*)d_in[4];
  const float* Wk = (const float*)d_in[5];
  const float* bk = (const float*)d_in[6];
  const float* Wv = (const float*)d_in[7];
  const float* bv = (const float*)d_in[8];
  const float* Wo = (const float*)d_in[9];
  const float* bo = (const float*)d_in[10];
  float* out = (float*)d_out;

  const size_t NE = (size_t)2 * NHEAD * S_LEN * HDIM;  // 4M elems per buffer
  bf16* Qb = (bf16*)d_ws;
  bf16* Kb = Qb + NE;
  bf16* Vt = Kb + NE;
  bf16* Ob = Vt + NE;   // [B][S][D] bf16; total ws use = 32 MB

  hipLaunchKernelGGL(qkv_proj_kernel, dim3(16, 32, 3), dim3(256), 0, stream,
                     q, k, v, Wq, bq, Wk, bk, Wv, bv, Qb, Kb, Vt);
  hipLaunchKernelGGL(flash_attn_kernel, dim3(1024), dim3(128), 0, stream,
                     Qb, Kb, Vt, Ob);
  hipLaunchKernelGGL(out_proj_kernel, dim3(16, 32), dim3(256), 0, stream,
                     Ob, Wo, bo, out);

  (void)in_sizes; (void)n_in; (void)out_size; (void)ws_size;
}
